// MoMGRUCell_89515708383669
// MI455X (gfx1250) — matrix operand order repaired and checked by reading, not compiled, not run on hardware
//
#include <hip/hip_runtime.h>
#include <hip/hip_bf16.h>
#include <cstdint>

// ---------------------------------------------------------------------------
// MoM-GRU cell for MI455X (gfx1250, wave32, WMMA bf16 16x16x32 + TDM/async-LDS)
// B=2048, I=1024, H=1024, M=8, S=128, D=64, RH=32
// ---------------------------------------------------------------------------

#define Bsz  2048
#define Idim 1024
#define Hdim 1024
#define Mmem 8
#define Sslt 128
#define Ddim 64
#define RHd  32
#define KXH  (Idim + Hdim)          // 2048
#define KGATE (Idim + Ddim + Hdim)  // 2112
#define SCALE 0.125f                // 1/sqrt(64)

typedef __attribute__((ext_vector_type(16))) __bf16 bf16x16;
typedef __attribute__((ext_vector_type(8)))  __bf16 bf16x8;
typedef __attribute__((ext_vector_type(8)))  float  f32x8;
typedef __attribute__((ext_vector_type(4)))  uint32_t u32x4;
typedef __attribute__((ext_vector_type(8)))  uint32_t u32x8;

// ---------------------------------------------------------------------------
// WMMA tile loads, contiguous k order (not the ISA table order; A and B agree):
//  A 16x32 bf16: lane L holds row M=L%16; half=L/16:
//    elements e0..15 -> K = k0 + e + 16*half   (one contiguous 32B run)
//  B 32x16 bf16 (pre-transposed weights Wt[n][k]): lane L holds col N=L%16;
//    elements e0..15 -> K = k0 + e + 16*half   (one contiguous 32B run)
// ---------------------------------------------------------------------------
__device__ __forceinline__ bf16x16 mom_load_a(const __bf16* p, int half) {
    union { bf16x16 v; bf16x8 h[2]; } u;
    u.h[0] = *(const bf16x8*)(p + 16 * half);
    u.h[1] = *(const bf16x8*)(p + 16 * half + 8);
    return u.v;
}
__device__ __forceinline__ bf16x16 mom_load_b(const __bf16* p, int half) {
    const __bf16* q = p + 16 * half;
    union { bf16x16 v; bf16x8 h[2]; } u;
    u.h[0] = *(const bf16x8*)(q);
    u.h[1] = *(const bf16x8*)(q + 8);
    return u.v;
}

__device__ __forceinline__ void mom_mma_seg(const __bf16* A, int lda, int K,
                                            const __bf16* Wt, int ldw, int kbase,
                                            int aRow, int bCol0, int half,
                                            f32x8& acc0, f32x8& acc1) {
    for (int k0 = 0; k0 < K; k0 += 32) {
        bf16x16 a  = mom_load_a(A + (size_t)aRow * lda + k0, half);
        bf16x16 b0 = mom_load_b(Wt + (size_t)bCol0 * ldw + kbase + k0, half);
        bf16x16 b1 = mom_load_b(Wt + (size_t)(bCol0 + 16) * ldw + kbase + k0, half);
        acc0 = __builtin_amdgcn_wmma_f32_16x16x32_bf16(false, a, false, b0,
                                                       (short)0, acc0, false, false);
        acc1 = __builtin_amdgcn_wmma_f32_16x16x32_bf16(false, a, false, b1,
                                                       (short)0, acc1, false, false);
    }
}

// ---------------------------------------------------------------------------
// Generic WMMA GEMM: out[B, N] = [A1 | A2 | A3] @ Wt^T (+bias, epilogue)
// mode 0: outF = acc + bias
// mode 1: outF = sigmoid(acc + bias)
// mode 2: outB = bf16( sigmoid(acc + bias) * extra[row,col] )   (r*hidden)
// ---------------------------------------------------------------------------
__global__ void mom_wmma_gemm(const __bf16* A1, int K1,
                              const __bf16* A2, int K2,
                              const __bf16* A3, int K3,
                              const __bf16* Wt, int Ktot, int N,
                              const float* biasA, const float* biasB, int nsplit,
                              const float* extra,
                              float* outF, __bf16* outB, int mode) {
    const int lane = threadIdx.x & 31;
    const int wave = threadIdx.x >> 5;
    const int half = lane >> 4;
    const int mBase = (blockIdx.x * 8 + wave) * 16;
    const int nBase = blockIdx.y * 32;
    const int aRow  = mBase + (lane & 15);
    const int bCol0 = nBase + (lane & 15);

    f32x8 acc0 = {}; f32x8 acc1 = {};
    int kbase = 0;
    if (K1 > 0) { mom_mma_seg(A1, K1, K1, Wt, Ktot, kbase, aRow, bCol0, half, acc0, acc1); kbase += K1; }
    if (K2 > 0) { mom_mma_seg(A2, K2, K2, Wt, Ktot, kbase, aRow, bCol0, half, acc0, acc1); kbase += K2; }
    if (K3 > 0) { mom_mma_seg(A3, K3, K3, Wt, Ktot, kbase, aRow, bCol0, half, acc0, acc1); kbase += K3; }

    #pragma unroll
    for (int i = 0; i < 8; ++i) {
        const int row = mBase + i + 8 * half;
        #pragma unroll
        for (int j = 0; j < 2; ++j) {
            const int col = nBase + (lane & 15) + 16 * j;
            float v = (j == 0) ? acc0[i] : acc1[i];
            v += (col < nsplit) ? biasA[col] : biasB[col - nsplit];
            const size_t idx = (size_t)row * N + col;
            if (mode == 0) {
                outF[idx] = v;
            } else if (mode == 1) {
                outF[idx] = 1.f / (1.f + __expf(-v));
            } else { // mode 2
                const float s = 1.f / (1.f + __expf(-v));
                outB[idx] = (__bf16)(s * extra[idx]);
            }
        }
    }
}

// ---------------------------------------------------------------------------
// bf16 conversion / weight packing
// ---------------------------------------------------------------------------
__global__ void mom_cvt_bf16(const float* __restrict__ x, __bf16* __restrict__ y, int n) {
    int i = blockIdx.x * blockDim.x + threadIdx.x;
    if (i < n) y[i] = (__bf16)x[i];
}

__global__ void mom_pack_cat96(const float* rW1, const float* rqW, __bf16* Wt) {
    int k = blockIdx.x * blockDim.x + threadIdx.x;   // 0..2047
    int n = blockIdx.y;                              // 0..95
    float v = (n < 32) ? rW1[k * RHd + n] : rqW[k * Ddim + (n - 32)];
    Wt[(size_t)n * KXH + k] = (__bf16)v;
}

__global__ void mom_pack_gate(const float* Wx, const float* Wh, __bf16* Wt) {
    int k = blockIdx.x * blockDim.x + threadIdx.x;
    int n = blockIdx.y;
    if (k >= KGATE) return;
    float v = (k < Idim + Ddim) ? Wx[(size_t)k * Hdim + n]
                                : Wh[(size_t)(k - Idim - Ddim) * Hdim + n];
    Wt[(size_t)n * KGATE + k] = (__bf16)v;
}

__global__ void mom_pack_kv(const float* wkW, const float* wvW, __bf16* Wt) {
    int k = blockIdx.x * blockDim.x + threadIdx.x;   // 0..1023
    int n = blockIdx.y;                              // 0..127
    float v = (n < 64) ? wkW[k * Ddim + n] : wvW[k * Ddim + (n - 64)];
    Wt[(size_t)n * Hdim + k] = (__bf16)v;
}

// ---------------------------------------------------------------------------
// Router finish. blockDim=64, grid=B.
// ---------------------------------------------------------------------------
__global__ void mom_router_finish(const float* out96, const float* rW2, const float* rb2,
                                  float* router_w, float* query) {
    __shared__ float r1[RHd];
    __shared__ float lg[Mmem];
    const int b = blockIdx.x, t = threadIdx.x;
    if (t < RHd) r1[t] = fmaxf(out96[(size_t)b * 96 + t], 0.f);
    __syncthreads();
    if (t < Mmem) {
        float a = rb2[t];
        #pragma unroll
        for (int k = 0; k < RHd; ++k) a += r1[k] * rW2[k * Mmem + t];
        lg[t] = a;
    }
    __syncthreads();
    if (t == 0) {
        float mx = lg[0];
        #pragma unroll
        for (int m = 1; m < Mmem; ++m) mx = fmaxf(mx, lg[m]);
        float s = 0.f, e[Mmem];
        #pragma unroll
        for (int m = 0; m < Mmem; ++m) { e[m] = __expf(lg[m] - mx); s += e[m]; }
        const float inv = 1.f / s;
        #pragma unroll
        for (int m = 0; m < Mmem; ++m) router_w[b * Mmem + m] = e[m] * inv;
    }
    query[(size_t)b * Ddim + t] = out96[(size_t)b * 96 + 32 + t];
}

// ---------------------------------------------------------------------------
// Read attention: one block per (b,m). 128 threads.
// Tile staged via the Tensor Data Mover: one descriptor copies the 32KB
// [128,64] f32 tile global->LDS. D# padding (interval=64 DWORDs, amount=1)
// reproduces the bank-conflict-free stride-65 LDS layout in hardware.
// (CDNA5 ISA; completion via TENSORcnt.)
// ---------------------------------------------------------------------------
#define LT 65
__global__ void mom_read_attn(const float* __restrict__ mem, const float* __restrict__ query,
                              const float* __restrict__ rw, float* __restrict__ ctxw) {
    __shared__ float q[Ddim];
    __shared__ float tile[Sslt * LT];
    __shared__ float red[Sslt];
    __shared__ float attn[Sslt];
    const int b = blockIdx.x, m = blockIdx.y, t = threadIdx.x;
    const size_t base = ((size_t)(b * Mmem + m)) * Sslt * Ddim;

    if (t < Ddim) q[t] = query[(size_t)b * Ddim + t];

    if (t < 32) {  // wave 0 issues the TDM descriptor (uniform -> SGPRs)
        const uint64_t ga = (uint64_t)(uintptr_t)(mem + base);
        const uint32_t galo = __builtin_amdgcn_readfirstlane((uint32_t)ga);
        const uint32_t gahi = __builtin_amdgcn_readfirstlane((uint32_t)(ga >> 32));
        const uint32_t ldsa = __builtin_amdgcn_readfirstlane((uint32_t)(uintptr_t)&tile[0]);
        u32x4 g0;
        g0[0] = 1u;                                    // count=1 (valid user D#)
        g0[1] = ldsa;                                  // lds_addr (bytes)
        g0[2] = galo;                                  // global_addr[31:0]
        g0[3] = (gahi & 0x01FFFFFFu) | (2u << 30);     // global_addr[56:32] | type=2
        u32x8 g1;
        g1[0] = (2u << 16) | (1u << 20) | (5u << 22);  // data_size=4B, pad_en, ivl=64dw, amt=1dw
        g1[1] = (uint32_t)(Sslt * Ddim) << 16;         // tensor_dim0[15:0]=8192 (bits 79:48)
        g1[2] = 1u << 16;                              // tensor_dim1=1 (bits 111:80)
        g1[3] = (uint32_t)(Sslt * Ddim) << 16;         // tile_dim0=8192 (bits 127:112)
        g1[4] = 1u;                                    // tile_dim1=1, tile_dim2=0
        g1[5] = (uint32_t)(Sslt * Ddim);               // tensor_dim0_stride (bits 207:160)
        g1[6] = 0u;
        g1[7] = 0u;                                    // tensor_dim1_stride=0
        asm volatile("tensor_load_to_lds %0, %1" :: "s"(g0), "s"(g1) : "memory");
        __builtin_amdgcn_s_wait_tensorcnt(0);
    }
    __syncthreads();

    float sc = 0.f;
    #pragma unroll 8
    for (int d = 0; d < Ddim; ++d) sc += q[d] * tile[t * LT + d];
    sc *= SCALE;
    red[t] = sc; __syncthreads();
    for (int o = 64; o > 0; o >>= 1) { if (t < o) red[t] = fmaxf(red[t], red[t + o]); __syncthreads(); }
    const float mx = red[0]; __syncthreads();
    const float e = __expf(sc - mx);
    red[t] = e; __syncthreads();
    for (int o = 64; o > 0; o >>= 1) { if (t < o) red[t] += red[t + o]; __syncthreads(); }
    const float inv = 1.f / red[0]; __syncthreads();
    attn[t] = e * inv; __syncthreads();
    const int d = t & 63, hf = t >> 6;
    float acc = 0.f;
    for (int s = hf * 64; s < hf * 64 + 64; ++s) acc += attn[s] * tile[s * LT + d];
    red[t] = acc; __syncthreads();
    if (t < Ddim)
        ctxw[((size_t)(b * Mmem + m)) * Ddim + t] = (red[t] + red[t + 64]) * rw[b * Mmem + m];
}

__global__ void mom_ctx_reduce(const float* __restrict__ ctxw, __bf16* __restrict__ ctxb) {
    int i = blockIdx.x * blockDim.x + threadIdx.x;   // over B*64
    int b = i >> 6, d = i & 63;
    float a = 0.f;
    #pragma unroll
    for (int m = 0; m < Mmem; ++m) a += ctxw[((size_t)(b * Mmem + m)) * Ddim + d];
    ctxb[i] = (__bf16)a;
}

__global__ void mom_gru_combine(const float* __restrict__ z, const float* __restrict__ ph,
                                const float* __restrict__ hid,
                                float* __restrict__ out0, float* __restrict__ out1,
                                __bf16* __restrict__ nhb) {
    int i = blockIdx.x * blockDim.x + threadIdx.x;   // over B*H
    float zz = z[i];
    float nh = (1.f - zz) * hid[i] + zz * tanhf(ph[i]);
    out0[i] = nh; out1[i] = nh; nhb[i] = (__bf16)nh;
}

__global__ void mom_wgate(const float* __restrict__ nh, const float* __restrict__ wgW,
                          const float* __restrict__ wgb, float* __restrict__ wgate) {
    __shared__ float red[128];
    const int b = blockIdx.x, t = threadIdx.x;
    float a = 0.f;
    for (int k = t; k < Hdim; k += 128) a += nh[(size_t)b * Hdim + k] * wgW[k];
    red[t] = a; __syncthreads();
    for (int o = 64; o > 0; o >>= 1) { if (t < o) red[t] += red[t + o]; __syncthreads(); }
    if (t == 0) wgate[b] = 1.f / (1.f + __expf(-(red[0] + wgb[0])));
}

// ---------------------------------------------------------------------------
// Write attention + memory update: one block per (b,m). 128 threads.
// Tile staged with per-lane async LDS loads (global_load_async_to_lds_b128,
// ASYNCcnt) -- no VGPR round trip. Padded LDS rows written 16B at a time
// (each 16B chunk stays inside one row, so chunks are LDS-contiguous).
// ---------------------------------------------------------------------------
__global__ void mom_write_update(const float* __restrict__ mem, const float* __restrict__ kv,
                                 const float* __restrict__ rw, const float* __restrict__ wgate,
                                 float* __restrict__ outMem) {
    __shared__ float key[Ddim], er[Ddim], ad[Ddim];
    __shared__ float tile[Sslt * LT];
    __shared__ float red[Sslt];
    __shared__ float attn[Sslt];
    const int b = blockIdx.x, m = blockIdx.y, t = threadIdx.x;
    const size_t base = ((size_t)(b * Mmem + m)) * Sslt * Ddim;
    const float4* g = (const float4*)(mem + base);

    #pragma unroll 4
    for (int i = t; i < Sslt * Ddim / 4; i += 128) {
        int s = i >> 4, d4 = (i & 15) << 2;
        uint32_t ldsb = (uint32_t)(uintptr_t)(&tile[s * LT + d4]);
        uint64_t ga   = (uint64_t)(uintptr_t)(g + i);
        asm volatile("global_load_async_to_lds_b128 %0, %1, off"
                     :: "v"(ldsb), "v"(ga) : "memory");
    }
    if (t < Ddim) {
        key[t] = kv[(size_t)b * 128 + t];
        float v = kv[(size_t)b * 128 + 64 + t];
        er[t] = 1.f / (1.f + __expf(-v));
        ad[t] = tanhf(v);
    }
    asm volatile("s_wait_asynccnt 0x0" ::: "memory");
    __syncthreads();

    float sc = 0.f;
    #pragma unroll 8
    for (int d = 0; d < Ddim; ++d) sc += key[d] * tile[t * LT + d];
    sc *= SCALE;
    red[t] = sc; __syncthreads();
    for (int o = 64; o > 0; o >>= 1) { if (t < o) red[t] = fmaxf(red[t], red[t + o]); __syncthreads(); }
    const float mx = red[0]; __syncthreads();
    const float e = __expf(sc - mx);
    red[t] = e; __syncthreads();
    for (int o = 64; o > 0; o >>= 1) { if (t < o) red[t] += red[t + o]; __syncthreads(); }
    const float gg = rw[b * Mmem + m] * wgate[b] / red[0]; __syncthreads();
    attn[t] = e * gg;   // scaled_s
    __syncthreads();
    float4* o4 = (float4*)(outMem + base);
    for (int i = t; i < Sslt * Ddim / 4; i += 128) {
        int s = i >> 4, d4 = (i & 15) << 2;
        const float scs = attn[s];
        const float* row = &tile[s * LT + d4];
        float4 o;
        o.x = row[0] * (1.f - scs * er[d4 + 0]) + scs * ad[d4 + 0];
        o.y = row[1] * (1.f - scs * er[d4 + 1]) + scs * ad[d4 + 1];
        o.z = row[2] * (1.f - scs * er[d4 + 2]) + scs * ad[d4 + 2];
        o.w = row[3] * (1.f - scs * er[d4 + 3]) + scs * ad[d4 + 3];
        o4[i] = o;
    }
}

// ---------------------------------------------------------------------------
// Host side
// ---------------------------------------------------------------------------
extern "C" void kernel_launch(void* const* d_in, const int* in_sizes, int n_in,
                              void* d_out, int out_size, void* d_ws, size_t ws_size,
                              hipStream_t stream) {
    const float* input   = (const float*)d_in[0];
    const float* hidden  = (const float*)d_in[1];
    const float* mem     = (const float*)d_in[2];
    const float* W_xr    = (const float*)d_in[3];
    const float* W_hr    = (const float*)d_in[4];
    const float* b_r     = (const float*)d_in[5];
    const float* W_xz    = (const float*)d_in[6];
    const float* W_hz    = (const float*)d_in[7];
    const float* b_z     = (const float*)d_in[8];
    const float* W_xh    = (const float*)d_in[9];
    const float* W_hh    = (const float*)d_in[10];
    const float* b_h     = (const float*)d_in[11];
    const float* rW1     = (const float*)d_in[12];
    const float* rb1     = (const float*)d_in[13];
    const float* rW2     = (const float*)d_in[14];
    const float* rb2     = (const float*)d_in[15];
    const float* rqW     = (const float*)d_in[16];
    const float* rqb     = (const float*)d_in[17];
    const float* wkW     = (const float*)d_in[18];
    const float* wkb     = (const float*)d_in[19];
    const float* wvW     = (const float*)d_in[20];
    const float* wvb     = (const float*)d_in[21];
    const float* wgW     = (const float*)d_in[22];
    const float* wgb     = (const float*)d_in[23];

    float* out_h0  = (float*)d_out;                                  // [B,H]
    float* out_h1  = out_h0 + (size_t)Bsz * Hdim;                    // [B,H]
    float* out_mem = out_h1 + (size_t)Bsz * Hdim;                    // [B,M,S,D]

    char* wsc = (char*)d_ws;
    size_t off = 0;
    auto alloc = [&](size_t bytes) -> void* {
        void* p = wsc + off;
        off = (off + bytes + 255) & ~(size_t)255;
        return p;
    };
    float*  out96   = (float*) alloc((size_t)Bsz * 96 * 4);
    float*  routerw = (float*) alloc((size_t)Bsz * Mmem * 4);
    float*  query   = (float*) alloc((size_t)Bsz * Ddim * 4);
    float*  ctxw    = (float*) alloc((size_t)Bsz * Mmem * Ddim * 4);
    float*  zbuf    = (float*) alloc((size_t)Bsz * Hdim * 4);
    float*  prehb   = (float*) alloc((size_t)Bsz * Hdim * 4);
    float*  kv      = (float*) alloc((size_t)Bsz * 128 * 4);
    float*  wgate   = (float*) alloc((size_t)Bsz * 4);
    __bf16* xin_b   = (__bf16*)alloc((size_t)Bsz * Idim * 2);
    __bf16* hid_b   = (__bf16*)alloc((size_t)Bsz * Hdim * 2);
    __bf16* ctx_b   = (__bf16*)alloc((size_t)Bsz * Ddim * 2);
    __bf16* rh_b    = (__bf16*)alloc((size_t)Bsz * Hdim * 2);
    __bf16* nh_b    = (__bf16*)alloc((size_t)Bsz * Hdim * 2);
    __bf16* Wcat    = (__bf16*)alloc((size_t)96 * KXH * 2);
    __bf16* Wrt     = (__bf16*)alloc((size_t)Hdim * KGATE * 2);
    __bf16* Wzt     = (__bf16*)alloc((size_t)Hdim * KGATE * 2);
    __bf16* Wht     = (__bf16*)alloc((size_t)Hdim * KGATE * 2);
    __bf16* Wkvt    = (__bf16*)alloc((size_t)128 * Hdim * 2);
    (void)ws_size; (void)n_in; (void)in_sizes; (void)out_size;

    const int BH = Bsz * Hdim;

    // 1. bf16 conversions
    mom_cvt_bf16<<<BH / 256, 256, 0, stream>>>(input,  xin_b, BH);
    mom_cvt_bf16<<<BH / 256, 256, 0, stream>>>(hidden, hid_b, BH);

    // 2. weight packing (transposed bf16)
    mom_pack_cat96<<<dim3(KXH / 256, 96),   256, 0, stream>>>(rW1, rqW, Wcat);
    mom_pack_gate <<<dim3((KGATE + 255) / 256, Hdim), 256, 0, stream>>>(W_xr, W_hr, Wrt);
    mom_pack_gate <<<dim3((KGATE + 255) / 256, Hdim), 256, 0, stream>>>(W_xz, W_hz, Wzt);
    mom_pack_gate <<<dim3((KGATE + 255) / 256, Hdim), 256, 0, stream>>>(W_xh, W_hh, Wht);
    mom_pack_kv   <<<dim3(Hdim / 256, 128), 256, 0, stream>>>(wkW, wvW, Wkvt);

    // 3. router-hidden + query GEMM: [xin|hidden] @ Wcat  (K=2048, N=96)
    mom_wmma_gemm<<<dim3(Bsz / 128, 96 / 32), 256, 0, stream>>>(
        xin_b, Idim, nullptr, 0, hid_b, Hdim, Wcat, KXH, 96,
        rb1, rqb, 32, nullptr, out96, nullptr, 0);

    // 4. router softmax + query extract
    mom_router_finish<<<Bsz, 64, 0, stream>>>(out96, rW2, rb2, routerw, query);

    // 5. read attention over memory (TDM-staged tiles)
    mom_read_attn<<<dim3(Bsz, Mmem), 128, 0, stream>>>(mem, query, routerw, ctxw);
    mom_ctx_reduce<<<(Bsz * Ddim) / 256, 256, 0, stream>>>(ctxw, ctx_b);

    // 6. GRU gate GEMMs (K = input 1024 | context 64 | hidden 1024 = 2112)
    mom_wmma_gemm<<<dim3(Bsz / 128, Hdim / 32), 256, 0, stream>>>(
        xin_b, Idim, ctx_b, Ddim, hid_b, Hdim, Wrt, KGATE, Hdim,
        b_r, b_r, Hdim, hidden, nullptr, rh_b, 2);           // rh = sigmoid(.)*h (bf16)
    mom_wmma_gemm<<<dim3(Bsz / 128, Hdim / 32), 256, 0, stream>>>(
        xin_b, Idim, ctx_b, Ddim, hid_b, Hdim, Wzt, KGATE, Hdim,
        b_z, b_z, Hdim, nullptr, zbuf, nullptr, 1);          // z = sigmoid(.)
    mom_wmma_gemm<<<dim3(Bsz / 128, Hdim / 32), 256, 0, stream>>>(
        xin_b, Idim, ctx_b, Ddim, rh_b, Hdim, Wht, KGATE, Hdim,
        b_h, b_h, Hdim, nullptr, prehb, nullptr, 0);         // pre_h

    // 7. combine -> new_hidden (both d_out copies + bf16 for projections)
    mom_gru_combine<<<BH / 256, 256, 0, stream>>>(zbuf, prehb, hidden, out_h0, out_h1, nh_b);

    // 8. write projections: [wkey|wval] = nh @ [wk_W|wv_W]  (N=128)
    mom_wmma_gemm<<<dim3(Bsz / 128, 128 / 32), 256, 0, stream>>>(
        nh_b, Hdim, nullptr, 0, nullptr, 0, Wkvt, Hdim, 128,
        wkb, wvb, 64, nullptr, kv, nullptr, 0);
    mom_wgate<<<Bsz, 128, 0, stream>>>(out_h0, wgW, wgb, wgate);

    // 9. write attention + memory update (async-LDS staged, streams ~1 GB HBM)
    mom_write_update<<<dim3(Bsz, Mmem), 128, 0, stream>>>(mem, kv, routerw, wgate, out_mem);
}
